// ChildSumTreeLSTMWithGates_33792802685044
// MI455X (gfx1250) — compile-verified, matrix operand order checked by reading, not run on hardware
//
#include <hip/hip_runtime.h>

// ---------------------------------------------------------------------------
// Child-Sum TreeLSTM, complete binary tree. B=256, N=1023, D_IN=D_H=256.
// Level-by-level kernels; GEMMs via v_wmma_f32_16x16x32_f16 (f16 in, f32 acc).
// M-tile = 32 batch rows per workgroup (2 x 16-row WMMA tiles) so each weight
// B-fragment load feeds two WMMAs; U_f fragment shared by both child GEMMs.
// f-gate phase runs first and collapses to a 16-value partial sum so iou-phase
// accumulators never coexist with f-phase accumulators (no spills).
// ---------------------------------------------------------------------------

#define TREE_N   1023
#define DEPTH    9
#define D        256
#define BATCH    256
#define OUT_HALF 65536          // B * D

typedef __attribute__((ext_vector_type(16))) _Float16 v16h;
typedef __attribute__((ext_vector_type(8)))  _Float16 v8h;
typedef __attribute__((ext_vector_type(8)))  float    v8f;

// f16 weight workspace layout (element offsets)
#define OFF_WIOU 0                    // 768*256
#define OFF_UIOU 196608               // 768*256
#define OFF_WF   393216               // 256*256
#define OFF_UF   458752               // 256*256
#define W_ELEMS  524288

__device__ __forceinline__ float sigf(float v) {
    return 1.0f / (1.0f + __expf(-v));
}
__device__ __forceinline__ float tanh_fast(float v) {
    float a = fabsf(v);
    float t = __expf(-2.0f * a);
    float r = (1.0f - t) / (1.0f + t);
    return copysignf(r, v);
}

// WMMA A/B fragment loader for row-major f16 source with leading dim 256.
// ISA 16-bit A layout: lanes 0-15 = rows M0-15 (kh=0), lanes 16-31 same rows
// with K-halves 8..15 / 24..31 (kh=8). Elems 0-7 = K=k0+kh.., 8-15 = K=k0+16+kh..
__device__ __forceinline__ v16h load_frag(const _Float16* base, int k0, int lane) {
    int r  = lane & 15;
    int kh = (lane >> 4) << 3;
    const _Float16* p = base + r * 256 + k0 + kh;
    v8h lo = *(const v8h*)(p);
    v8h hi = *(const v8h*)(p + 16);
    return __builtin_shufflevector(lo, hi, 0,1,2,3,4,5,6,7,8,9,10,11,12,13,14,15);
}

#define WMMA_F16(a, b, c) \
    __builtin_amdgcn_wmma_f32_16x16x32_f16(false, (a), false, (b), (short)0, (c), false, false)

// ---------------------------------------------------------------------------
__global__ void tlstm_convert_weights(const float* __restrict__ Wiou,
                                      const float* __restrict__ Uiou,
                                      const float* __restrict__ Wf,
                                      const float* __restrict__ Uf,
                                      _Float16* __restrict__ dst) {
    int i = blockIdx.x * blockDim.x + threadIdx.x;
    if (i >= W_ELEMS) return;
    float v;
    if      (i < OFF_UIOU) v = Wiou[i - OFF_WIOU];
    else if (i < OFF_WF)   v = Uiou[i - OFF_UIOU];
    else if (i < OFF_UF)   v = Wf[i - OFF_WF];
    else                   v = Uf[i - OFF_UF];
    dst[i] = (_Float16)v;
}

// ---------------------------------------------------------------------------
// Leaf level: iou = x @ W_iou^T + b_iou ; c = sig(i)*tanh(u) ; h = sig(o)*tanh(c)
// One workgroup (256 thr = 8 waves) per (leaf node, 32-row batch tile-pair).
__global__ void __launch_bounds__(256, 2)
tlstm_leaf(const float* __restrict__ x,
           const float* __restrict__ b_iou,
           const _Float16* __restrict__ wbase,
           _Float16* __restrict__ h_h,
           float* __restrict__ c_f) {
    __shared__ __align__(16) _Float16 x_s[32 * 256];

    int node = (1 << DEPTH) - 1 + (blockIdx.x >> 3);
    int b0   = (blockIdx.x & 7) << 5;          // 32-row tile pair
    int lane = threadIdx.x & 31;
    int wave = threadIdx.x >> 5;

    for (int idx = threadIdx.x; idx < 32 * 256; idx += 256) {
        int r = idx >> 8, d = idx & 255;
        size_t gi = ((size_t)(b0 + r) * TREE_N + node) * D + d;
        x_s[idx] = (_Float16)x[gi];
    }
    __syncthreads();

    const _Float16* Wiou = wbase + OFF_WIOU;

    for (int gi2 = 0; gi2 < 2; ++gi2) {
        int g  = wave + gi2 * 8;               // 0..15
        int n0 = g << 4;
        const _Float16* bWi = Wiou + (size_t)(      n0) * 256;
        const _Float16* bWo = Wiou + (size_t)(256 + n0) * 256;
        const _Float16* bWu = Wiou + (size_t)(512 + n0) * 256;
        v8f ai0 = {}, ao0 = {}, au0 = {};
        v8f ai1 = {}, ao1 = {}, au1 = {};
        #pragma unroll
        for (int k0 = 0; k0 < 256; k0 += 32) {
            v16h ax0 = load_frag(x_s,            k0, lane);
            v16h ax1 = load_frag(x_s + 16 * 256, k0, lane);
            v16h b;
            b = load_frag(bWi, k0, lane);
            ai0 = WMMA_F16(ax0, b, ai0);  ai1 = WMMA_F16(ax1, b, ai1);
            b = load_frag(bWo, k0, lane);
            ao0 = WMMA_F16(ax0, b, ao0);  ao1 = WMMA_F16(ax1, b, ao1);
            b = load_frag(bWu, k0, lane);
            au0 = WMMA_F16(ax0, b, au0);  au1 = WMMA_F16(ax1, b, au1);
        }
        int ncol  = n0 + (lane & 15);
        float bi = b_iou[ncol], bo = b_iou[256 + ncol], bu = b_iou[512 + ncol];
        int mbase = (lane < 16) ? 0 : 8;
        #pragma unroll
        for (int t = 0; t < 2; ++t) {
            v8f ai = t ? ai1 : ai0, ao = t ? ao1 : ao0, au = t ? au1 : au0;
            #pragma unroll
            for (int j = 0; j < 8; ++j) {
                int b = b0 + t * 16 + mbase + j;
                float iv = sigf(ai[j] + bi);
                float ov = sigf(ao[j] + bo);
                float uv = tanh_fast(au[j] + bu);
                float cv = iv * uv;
                float hv = ov * tanh_fast(cv);
                size_t oidx = ((size_t)b * TREE_N + node) * D + ncol;
                c_f[oidx] = cv;
                h_h[oidx] = (_Float16)hv;
            }
        }
    }
}

// ---------------------------------------------------------------------------
// Internal level: full child-sum LSTM cell. One workgroup per (node, 32-row pair).
__global__ void __launch_bounds__(256, 2)
tlstm_internal(const float* __restrict__ x,
               const float* __restrict__ b_iou,
               const float* __restrict__ b_f,
               const _Float16* __restrict__ wbase,
               _Float16* __restrict__ h_h,
               float* __restrict__ c_f,
               float* __restrict__ out,
               int level_start, int write_out) {
    __shared__ __align__(16) _Float16 x_s [32 * 256];
    __shared__ __align__(16) _Float16 hl_s[32 * 256];
    __shared__ __align__(16) _Float16 hr_s[32 * 256];
    __shared__ __align__(16) _Float16 hs_s[32 * 256];

    int node = level_start + (blockIdx.x >> 3);
    int b0   = (blockIdx.x & 7) << 5;
    int nl   = 2 * node + 1;
    int nr   = 2 * node + 2;
    int lane = threadIdx.x & 31;
    int wave = threadIdx.x >> 5;

    for (int idx = threadIdx.x; idx < 32 * 256; idx += 256) {
        int r = idx >> 8, d = idx & 255;
        size_t brow = (size_t)(b0 + r) * TREE_N;
        x_s[idx] = (_Float16)x[(brow + node) * D + d];
        _Float16 hl = h_h[(brow + nl) * D + d];
        _Float16 hr = h_h[(brow + nr) * D + d];
        hl_s[idx] = hl;
        hr_s[idx] = hr;
        hs_s[idx] = (_Float16)((float)hl + (float)hr);
    }
    __syncthreads();

    const _Float16* Wiou = wbase + OFF_WIOU;
    const _Float16* Uiou = wbase + OFF_UIOU;
    const _Float16* Wf   = wbase + OFF_WF;
    const _Float16* Uf   = wbase + OFF_UF;

    for (int gi2 = 0; gi2 < 2; ++gi2) {
        int g  = wave + gi2 * 8;               // 0..15
        int n0 = g << 4;
        int ncol  = n0 + (lane & 15);
        int mbase = (lane < 16) ? 0 : 8;

        // ---- Phase 1: fx = x@W_f^T ; fl/fr = h_{l,r}@U_f^T (shared U_f frag) ----
        {
            const _Float16* bWF = Wf + (size_t)n0 * 256;
            const _Float16* bUF = Uf + (size_t)n0 * 256;
            v8f afx0 = {}, afl0 = {}, afr0 = {};
            v8f afx1 = {}, afl1 = {}, afr1 = {};
            #pragma unroll
            for (int k0 = 0; k0 < 256; k0 += 32) {
                v16h ax0 = load_frag(x_s,             k0, lane);
                v16h ax1 = load_frag(x_s  + 16 * 256, k0, lane);
                v16h al0 = load_frag(hl_s,            k0, lane);
                v16h al1 = load_frag(hl_s + 16 * 256, k0, lane);
                v16h ar0 = load_frag(hr_s,            k0, lane);
                v16h ar1 = load_frag(hr_s + 16 * 256, k0, lane);
                v16h b;
                b = load_frag(bWF, k0, lane);
                afx0 = WMMA_F16(ax0, b, afx0);  afx1 = WMMA_F16(ax1, b, afx1);
                b = load_frag(bUF, k0, lane);   // one fragment feeds 4 WMMAs
                afl0 = WMMA_F16(al0, b, afl0);  afl1 = WMMA_F16(al1, b, afl1);
                afr0 = WMMA_F16(ar0, b, afr0);  afr1 = WMMA_F16(ar1, b, afr1);
            }
            // Collapse to fsum = sig(fx+bf+fl)*cl + sig(fx+bf+fr)*cr
            float bf = b_f[ncol];
            v8f fs0, fs1;
            #pragma unroll
            for (int j = 0; j < 8; ++j) {
                {
                    int b = b0 + mbase + j;
                    size_t brow = (size_t)b * TREE_N;
                    float cl = c_f[(brow + nl) * D + ncol];
                    float cr = c_f[(brow + nr) * D + ncol];
                    float fxv = afx0[j] + bf;
                    fs0[j] = sigf(fxv + afl0[j]) * cl + sigf(fxv + afr0[j]) * cr;
                }
                {
                    int b = b0 + 16 + mbase + j;
                    size_t brow = (size_t)b * TREE_N;
                    float cl = c_f[(brow + nl) * D + ncol];
                    float cr = c_f[(brow + nr) * D + ncol];
                    float fxv = afx1[j] + bf;
                    fs1[j] = sigf(fxv + afl1[j]) * cl + sigf(fxv + afr1[j]) * cr;
                }
            }

            // ---- Phase 2: iou = x@W_iou^T + h_sum@U_iou^T (both M-tiles) ----
            const _Float16* bWi = Wiou + (size_t)(      n0) * 256;
            const _Float16* bWo = Wiou + (size_t)(256 + n0) * 256;
            const _Float16* bWu = Wiou + (size_t)(512 + n0) * 256;
            const _Float16* bUi = Uiou + (size_t)(      n0) * 256;
            const _Float16* bUo = Uiou + (size_t)(256 + n0) * 256;
            const _Float16* bUu = Uiou + (size_t)(512 + n0) * 256;
            v8f ai0 = {}, ao0 = {}, au0 = {};
            v8f ai1 = {}, ao1 = {}, au1 = {};
            #pragma unroll
            for (int k0 = 0; k0 < 256; k0 += 32) {
                v16h ax0 = load_frag(x_s,             k0, lane);
                v16h ax1 = load_frag(x_s  + 16 * 256, k0, lane);
                v16h ah0 = load_frag(hs_s,            k0, lane);
                v16h ah1 = load_frag(hs_s + 16 * 256, k0, lane);
                v16h b;
                b = load_frag(bWi, k0, lane);
                ai0 = WMMA_F16(ax0, b, ai0);  ai1 = WMMA_F16(ax1, b, ai1);
                b = load_frag(bUi, k0, lane);
                ai0 = WMMA_F16(ah0, b, ai0);  ai1 = WMMA_F16(ah1, b, ai1);
                b = load_frag(bWo, k0, lane);
                ao0 = WMMA_F16(ax0, b, ao0);  ao1 = WMMA_F16(ax1, b, ao1);
                b = load_frag(bUo, k0, lane);
                ao0 = WMMA_F16(ah0, b, ao0);  ao1 = WMMA_F16(ah1, b, ao1);
                b = load_frag(bWu, k0, lane);
                au0 = WMMA_F16(ax0, b, au0);  au1 = WMMA_F16(ax1, b, au1);
                b = load_frag(bUu, k0, lane);
                au0 = WMMA_F16(ah0, b, au0);  au1 = WMMA_F16(ah1, b, au1);
            }

            // ---- Final gating ----
            float bi = b_iou[ncol], bo = b_iou[256 + ncol], bu = b_iou[512 + ncol];
            #pragma unroll
            for (int t = 0; t < 2; ++t) {
                v8f ai = t ? ai1 : ai0, ao = t ? ao1 : ao0, au = t ? au1 : au0;
                v8f fs = t ? fs1 : fs0;
                #pragma unroll
                for (int j = 0; j < 8; ++j) {
                    int b = b0 + t * 16 + mbase + j;
                    size_t brow = (size_t)b * TREE_N;
                    float iv = sigf(ai[j] + bi);
                    float ov = sigf(ao[j] + bo);
                    float uv = tanh_fast(au[j] + bu);
                    float cv = iv * uv + fs[j];
                    float hv = ov * tanh_fast(cv);
                    size_t oidx = (brow + node) * D + ncol;
                    c_f[oidx] = cv;
                    h_h[oidx] = (_Float16)hv;
                    if (write_out) {
                        out[(size_t)b * D + ncol]            = hv;
                        out[OUT_HALF + (size_t)b * D + ncol] = cv;
                    }
                }
            }
        }
    }
}

// ---------------------------------------------------------------------------
extern "C" void kernel_launch(void* const* d_in, const int* in_sizes, int n_in,
                              void* d_out, int out_size, void* d_ws, size_t ws_size,
                              hipStream_t stream) {
    (void)in_sizes; (void)n_in; (void)out_size; (void)ws_size;

    const float* x     = (const float*)d_in[0];
    const float* W_iou = (const float*)d_in[1];
    const float* b_iou = (const float*)d_in[2];
    const float* U_iou = (const float*)d_in[3];
    const float* W_f   = (const float*)d_in[4];
    const float* b_f   = (const float*)d_in[5];
    const float* U_f   = (const float*)d_in[6];
    float*       out   = (float*)d_out;

    // workspace: [ f16 weights 1MB | h (f16, B*N*D) | c (f32, B*N*D) ]
    _Float16* wbase = (_Float16*)d_ws;
    char*     p     = (char*)d_ws + (size_t)W_ELEMS * 2;
    _Float16* h_h   = (_Float16*)p;
    p += (size_t)BATCH * TREE_N * D * 2;
    float*    c_f   = (float*)p;

    tlstm_convert_weights<<<(W_ELEMS + 255) / 256, 256, 0, stream>>>(
        W_iou, U_iou, W_f, U_f, wbase);

    // leaves: 512 nodes x 8 batch tile-pairs
    tlstm_leaf<<<512 * 8, 256, 0, stream>>>(x, b_iou, wbase, h_h, c_f);

    // internal levels, bottom-up
    for (int l = DEPTH - 1; l >= 0; --l) {
        int s = (1 << l) - 1;
        int n = 1 << l;
        tlstm_internal<<<n * 8, 256, 0, stream>>>(
            x, b_iou, b_f, wbase, h_h, c_f, out, s, (l == 0) ? 1 : 0);
    }
}
